// Reconstruction_84911503442527
// MI455X (gfx1250) — matrix operand order repaired and checked by reading, not compiled, not run on hardware
//
#include <hip/hip_runtime.h>
#include <math.h>

// ---------------------------------------------------------------------------
// Problem constants (from reference): B=8, N=1024, FH=128 -> FIN=256, H=8, FO=64
// ---------------------------------------------------------------------------
#define B_   8
#define N_   1024
#define H_   8
#define FO_  64
#define FIN_ 256
#define SLOPE_ 0.2f
#define NEG_  (-1e9f)

typedef __attribute__((ext_vector_type(8)))  float  v8f;
typedef __attribute__((ext_vector_type(8)))  int    v8i;
typedef __attribute__((ext_vector_type(8)))  __bf16 v8bf;
typedef __attribute__((ext_vector_type(16))) __bf16 v16bf;

__device__ __forceinline__ v16bf cat16(v8bf lo, v8bf hi) {
    return __builtin_shufflevector(lo, hi, 0,1,2,3,4,5,6,7,8,9,10,11,12,13,14,15);
}

// ---------------------------------------------------------------------------
// Kernel 1: recx_bf16[b,n,f] = bf16( x[b,f] * Npost[n,f] * e_atten[b,n,f] )
// ---------------------------------------------------------------------------
__global__ void __launch_bounds__(256)
prep_recx_kernel(const float* __restrict__ x, const float* __restrict__ Npost,
                 const float* __restrict__ e_atten, __bf16* __restrict__ recx) {
    int idx = blockIdx.x * 256 + threadIdx.x;
    if (idx >= B_ * N_ * FIN_) return;
    int f   = idx & (FIN_ - 1);
    int bn  = idx / FIN_;
    int n   = bn & (N_ - 1);
    int b   = bn / N_;
    float v = x[b * FIN_ + f] * Npost[n * FIN_ + f] * e_atten[idx];
    recx[idx] = (__bf16)v;
}

// ---------------------------------------------------------------------------
// Kernel 2: wT_bf16[h,o,f] = bf16( w[h,f,o] )   (K-contiguous B operand)
// ---------------------------------------------------------------------------
__global__ void __launch_bounds__(256)
prep_wT_kernel(const float* __restrict__ w, __bf16* __restrict__ wT) {
    int idx = blockIdx.x * 256 + threadIdx.x;
    if (idx >= H_ * FO_ * FIN_) return;
    int f  = idx & (FIN_ - 1);
    int ho = idx / FIN_;
    int o  = ho & (FO_ - 1);
    int h  = ho / FO_;
    wT[idx] = (__bf16)w[(h * FIN_ + f) * FO_ + o];
}

// ---------------------------------------------------------------------------
// Kernel 3: h = recx @ w via v_wmma_f32_16x16x32_bf16.
// One wave -> 16 graph-rows x 64 FO cols for one (b,h). K loop: 256 in 8x32.
// Output stored transposed bf16: ht[b,h,o,n]  (K-contiguous for pass 2).
// ---------------------------------------------------------------------------
__global__ void __launch_bounds__(256)
gemm1_kernel(const __bf16* __restrict__ recx, const __bf16* __restrict__ wT,
             __bf16* __restrict__ ht) {
    const int gw   = blockIdx.x * 8 + (threadIdx.x >> 5);   // 4096 waves
    const int lane = threadIdx.x & 31;
    const int half = lane >> 4;        // lane-half per WMMA layout
    const int r0   = lane & 15;        // A: row, B/D: column
    const int bh   = gw >> 6;          // 64 row-tiles per (b,h)
    const int rt   = gw & 63;
    const int b    = bh >> 3;
    const int h    = bh & 7;
    const int rowbase = rt * 16;

    const v8f zero8 = {0.f,0.f,0.f,0.f,0.f,0.f,0.f,0.f};
    v8f acc[4];
#pragma unroll
    for (int t = 0; t < 4; ++t) acc[t] = zero8;

    const __bf16* arow = recx + (size_t)(b * N_ + rowbase + r0) * FIN_;

#pragma unroll
    for (int ks = 0; ks < 8; ++ks) {
        const int kb = ks * 32;
        // A 16x32 bf16: element e<8 -> K = kb + half*8 + e ; e>=8 -> +16
        v8bf alo = *(const v8bf*)(arow + kb + half * 8);
        v8bf ahi = *(const v8bf*)(arow + kb + 16 + half * 8);
        v16bf av = cat16(alo, ahi);
#pragma unroll
        for (int t = 0; t < 4; ++t) {
            const int o = t * 16 + r0;
            // B 32x16 bf16: element e<8 -> K = kb + half*8 + e ; e>=8 -> +16, col = r0
            const __bf16* wp = wT + (size_t)(h * FO_ + o) * FIN_ + kb + half * 8;
            v16bf bv = cat16(*(const v8bf*)wp, *(const v8bf*)(wp + 16));
            acc[t] = __builtin_amdgcn_wmma_f32_16x16x32_bf16(
                false, av, false, bv, (short)0, acc[t], false, false);
        }
    }
    // D 16x16 f32: element r -> (row = half*8 + r, col = r0); store transposed
#pragma unroll
    for (int t = 0; t < 4; ++t) {
        const int o = t * 16 + r0;
#pragma unroll
        for (int r = 0; r < 8; ++r) {
            ht[(size_t)(bh * FO_ + o) * N_ + rowbase + half * 8 + r] =
                (__bf16)acc[t][r];
        }
    }
}

// ---------------------------------------------------------------------------
// Kernel 4: s[b,h,n] = sum_o h[b,h,n,o]*a_src[h,o] ; same for d with a_dst
// ---------------------------------------------------------------------------
__global__ void __launch_bounds__(256)
sd_kernel(const __bf16* __restrict__ ht, const float* __restrict__ a_src,
          const float* __restrict__ a_dst, float* __restrict__ s,
          float* __restrict__ d) {
    int idx = blockIdx.x * 256 + threadIdx.x;     // B*H*N = 65536
    if (idx >= B_ * H_ * N_) return;
    int n  = idx & (N_ - 1);
    int bh = idx / N_;
    int h  = bh & (H_ - 1);
    float ss = 0.f, dd = 0.f;
#pragma unroll
    for (int o = 0; o < FO_; ++o) {
        float hv = (float)ht[(size_t)(bh * FO_ + o) * N_ + n];
        ss += hv * a_src[h * FO_ + o];
        dd += hv * a_dst[h * FO_ + o];
    }
    s[idx] = ss;
    d[idx] = dd;
}

// ---------------------------------------------------------------------------
// Kernel 5: fused masked-softmax + attn@h (flash-style online softmax).
// One wave -> 16 rows x FO=64 for one (b,h); m-loop: 1024 in 32x32.
// Probs packed to bf16 as the WMMA A operand; atomicAdd per-head into accum.
// ---------------------------------------------------------------------------
__global__ void __launch_bounds__(256)
attn_kernel(const int* __restrict__ A, const float* __restrict__ s,
            const float* __restrict__ d, const __bf16* __restrict__ ht,
            float* __restrict__ accum) {
    const int gw   = blockIdx.x * 8 + (threadIdx.x >> 5);   // 4096 waves
    const int lane = threadIdx.x & 31;
    const int half = lane >> 4;
    const int r0   = lane & 15;
    const int bh   = gw >> 6;
    const int rt   = gw & 63;
    const int b    = bh >> 3;
    const int rowbase = rt * 16;

    const float srow = s[bh * N_ + rowbase + r0];
    const float* dp  = d + bh * N_;
    const int* arow  = A + ((size_t)b * N_ + rowbase + r0) * N_;

    const v8f zero8 = {0.f,0.f,0.f,0.f,0.f,0.f,0.f,0.f};
    v8f acc[4];
#pragma unroll
    for (int t = 0; t < 4; ++t) acc[t] = zero8;
    float mrow = -3.0e38f;
    float lrow = 0.f;

    for (int mt = 0; mt < N_ / 32; ++mt) {
        const int mbase = mt * 32;
        // lane's 16 logit columns: e<8 -> m = mbase + half*8 + e ; e>=8 -> +16
        v8f dlo = *(const v8f*)(dp + mbase + half * 8);
        v8f dhi = *(const v8f*)(dp + mbase + 16 + half * 8);
        v8i alo = *(const v8i*)(arow + mbase + half * 8);
        v8i ahi = *(const v8i*)(arow + mbase + 16 + half * 8);
        if (mt + 1 < N_ / 32)
            __builtin_prefetch(arow + mbase + 32, 0, 0);   // global_prefetch_b8

        float z[16];
#pragma unroll
        for (int i = 0; i < 8; ++i) {
            float t0 = srow + dlo[i];
            t0 = (t0 > 0.f) ? t0 : t0 * SLOPE_;
            z[i] = (alo[i] != 0) ? t0 : NEG_;
            float t1 = srow + dhi[i];
            t1 = (t1 > 0.f) ? t1 : t1 * SLOPE_;
            z[8 + i] = (ahi[i] != 0) ? t1 : NEG_;
        }
        float tmax = z[0];
#pragma unroll
        for (int i = 1; i < 16; ++i) tmax = fmaxf(tmax, z[i]);
        tmax = fmaxf(tmax, __shfl_xor(tmax, 16, 32));   // combine lane halves
        const float mnew  = fmaxf(mrow, tmax);
        const float scale = __expf(mrow - mnew);

        v16bf pa;
        float psum = 0.f;
#pragma unroll
        for (int i = 0; i < 16; ++i) {
            float pv = __expf(z[i] - mnew);
            psum += pv;
            pa[i] = (__bf16)pv;
        }
        psum += __shfl_xor(psum, 16, 32);
        lrow = lrow * scale + psum;
        mrow = mnew;

        // rescale accumulators: element r sits on output row half*8 + r,
        // whose scale lives on lane (half*8 + r) (rows are lane%16 in A layout)
#pragma unroll
        for (int r = 0; r < 8; ++r) {
            float sc = __shfl(scale, half * 8 + r, 32);
#pragma unroll
            for (int t = 0; t < 4; ++t) acc[t][r] *= sc;
        }

        // P(16x32) @ H(32x64): 4 WMMAs, B operand K-contiguous from ht[b,h,o,m]
#pragma unroll
        for (int t = 0; t < 4; ++t) {
            const int o = t * 16 + r0;
            const __bf16* hp = ht + (size_t)(bh * FO_ + o) * N_ + mbase + half * 8;
            v16bf bv = cat16(*(const v8bf*)hp, *(const v8bf*)(hp + 16));
            acc[t] = __builtin_amdgcn_wmma_f32_16x16x32_bf16(
                false, pa, false, bv, (short)0, acc[t], false, false);
        }
    }

    // normalize by row-sum and accumulate this head's contribution
#pragma unroll
    for (int t = 0; t < 4; ++t) {
        const int o = t * 16 + r0;
#pragma unroll
        for (int r = 0; r < 8; ++r) {
            float lr  = __shfl(lrow, half * 8 + r, 32);
            float val = acc[t][r] / lr;
            atomicAdd(accum + (size_t)(b * N_ + rowbase + half * 8 + r) * FO_ + o,
                      val);
        }
    }
}

// ---------------------------------------------------------------------------
// Kernel 6: out = (accum / H + bias[o]) * mask_zero[b,n]
// ---------------------------------------------------------------------------
__global__ void __launch_bounds__(256)
finalize_kernel(const float* __restrict__ accum, const float* __restrict__ bias,
                const float* __restrict__ mask, float* __restrict__ out) {
    int idx = blockIdx.x * 256 + threadIdx.x;     // B*N*FO = 524288
    if (idx >= B_ * N_ * FO_) return;
    int o  = idx & (FO_ - 1);
    int bn = idx / FO_;
    out[idx] = (accum[idx] * (1.f / H_) + bias[o]) * mask[bn];
}

// ---------------------------------------------------------------------------
// Host launcher
// ---------------------------------------------------------------------------
extern "C" void kernel_launch(void* const* d_in, const int* in_sizes, int n_in,
                              void* d_out, int out_size, void* d_ws, size_t ws_size,
                              hipStream_t stream) {
    const float* x       = (const float*)d_in[0];
    const int*   A       = (const int*)  d_in[1];
    const float* mask    = (const float*)d_in[2];
    // d_in[3] = p_atten (unused by the reference forward body)
    const float* e_atten = (const float*)d_in[4];
    const float* Npost   = (const float*)d_in[5];
    const float* w       = (const float*)d_in[6];
    const float* a_src   = (const float*)d_in[7];
    const float* a_dst   = (const float*)d_in[8];
    const float* bias    = (const float*)d_in[9];
    float* out = (float*)d_out;

    // workspace carving (bytes)
    char* ws = (char*)d_ws;
    size_t off = 0;
    __bf16* recx = (__bf16*)(ws + off); off += (size_t)B_ * N_ * FIN_ * 2;      // 4 MB
    __bf16* wT   = (__bf16*)(ws + off); off += (size_t)H_ * FO_ * FIN_ * 2;     // 256 KB
    __bf16* ht   = (__bf16*)(ws + off); off += (size_t)B_ * H_ * FO_ * N_ * 2;  // 8 MB
    float*  sbuf = (float*) (ws + off); off += (size_t)B_ * H_ * N_ * 4;        // 256 KB
    float*  dbuf = (float*) (ws + off); off += (size_t)B_ * H_ * N_ * 4;        // 256 KB
    float*  accum= (float*) (ws + off); off += (size_t)B_ * N_ * FO_ * 4;       // 2 MB

    // zero the head accumulator (stream-ordered, graph-capture safe)
    hipMemsetAsync(accum, 0, (size_t)B_ * N_ * FO_ * 4, stream);

    prep_recx_kernel<<<(B_ * N_ * FIN_ + 255) / 256, 256, 0, stream>>>(
        x, Npost, e_atten, recx);
    prep_wT_kernel<<<(H_ * FO_ * FIN_ + 255) / 256, 256, 0, stream>>>(w, wT);

    // 4096 waves = 512 blocks x 256 threads (8 waves/block)
    gemm1_kernel<<<512, 256, 0, stream>>>(recx, wT, ht);

    sd_kernel<<<(B_ * H_ * N_ + 255) / 256, 256, 0, stream>>>(
        ht, a_src, a_dst, sbuf, dbuf);

    attn_kernel<<<512, 256, 0, stream>>>(A, sbuf, dbuf, ht, accum);

    finalize_kernel<<<(B_ * N_ * FO_ + 255) / 256, 256, 0, stream>>>(
        accum, bias, mask, out);
}
